// StackGRUDecoder_12807592476790
// MI455X (gfx1250) — matrix operand order repaired and checked by reading, not compiled, not run on hardware
//
#include <hip/hip_runtime.h>
#include <hip/hip_bf16.h>

// ---------------------------------------------------------------------------
// StackGRUDecoder for MI455X (gfx1250, wave32, WMMA)
// Strategy: convert all weights to bf16 once (fits 192MB L2 -> ~91MB),
// then a host loop of 50 decode steps:
//   embed-gather -> GRU L0 (wmma bf16) -> GRU L1 (wmma bf16)
//   -> logits GEMM (wmma bf16, fused bias + argmax via packed u64 atomicMax)
//   -> argmax decode (feeds next step's gather)
// All state ping-pongs inside d_ws; deterministic; graph-capture safe.
// ---------------------------------------------------------------------------

typedef __attribute__((ext_vector_type(16))) __bf16 v16bf;
typedef __attribute__((ext_vector_type(8)))  __bf16 v8bf;
typedef __attribute__((ext_vector_type(8)))  float  v8f;

#define GRU_H   1024
#define GRU_B   32
#define VOCAB   32000
#define NSTEP   50      // num_steps is a device scalar; fixed per setup_inputs()
#define START_INDEX 1

// ---- helpers --------------------------------------------------------------

__device__ __forceinline__ unsigned short f2bf(float f) {
    unsigned u = __float_as_uint(f);
    unsigned r = (u + 0x7FFFu + ((u >> 16) & 1u)) >> 16;   // round-nearest-even
    return (unsigned short)r;
}

__device__ __forceinline__ unsigned f2ord(float f) {
    // monotone map float -> u32 (total order matching float compare)
    unsigned u = __float_as_uint(f);
    return (u & 0x80000000u) ? ~u : (u | 0x80000000u);
}

__device__ __forceinline__ float sigm(float x) {
    return 1.0f / (1.0f + __expf(-x));
}

__device__ __forceinline__ v8f wmma_bf16(v16bf a, v16bf b, v8f c) {
    // v_wmma_f32_16x16x32_bf16: D = A(16x32) * B(32x16) + C(16x16 f32)
    return __builtin_amdgcn_wmma_f32_16x16x32_bf16(
        /*neg_a=*/false, a, /*neg_b=*/false, b,
        /*c_mod=*/(short)0, c, /*reuse_a=*/false, /*reuse_b=*/false);
}

// A-matrix 16x32 bf16 fragment (contiguous K, same order as B): lane = M row (lane&15);
// elements [0..7]  = K = (lane>>4)*16 + 0..7
// elements [8..15] = K = (lane>>4)*16 + 8..15
__device__ __forceinline__ v16bf load_a_frag(const unsigned short* __restrict__ row,
                                             int kk, int abase) {
    v16bf a;
    *reinterpret_cast<v8bf*>(&a) =
        *reinterpret_cast<const v8bf*>(row + kk + abase);
    *(reinterpret_cast<v8bf*>(&a) + 1) =
        *reinterpret_cast<const v8bf*>(row + kk + abase + 8);
    return a;
}

// B-matrix 32x16 bf16 fragment: lane&15 = N column; elements 0..15 are
// contiguous K starting at (lane>>4)*16.  B[k][n] = W[n][k] (row-major W),
// so this is one contiguous 32B load from W's row n.
__device__ __forceinline__ v16bf load_b_frag(const unsigned short* __restrict__ wrow,
                                             int kk, int bbase) {
    return *reinterpret_cast<const v16bf*>(wrow + kk + bbase);
}

// ---- setup kernels --------------------------------------------------------

__global__ void cvt_bf16_kernel(const float* __restrict__ src,
                                unsigned short* __restrict__ dst, int n) {
    for (int i = blockIdx.x * blockDim.x + threadIdx.x; i < n;
         i += gridDim.x * blockDim.x)
        dst[i] = f2bf(src[i]);
}

__global__ void init_state_kernel(const float* __restrict__ hidden,   // [2,32,1024]
                                  float* __restrict__ h0f, float* __restrict__ h1f,
                                  unsigned short* __restrict__ h0b,
                                  unsigned short* __restrict__ h1b,
                                  int* __restrict__ last,
                                  unsigned long long* __restrict__ packed) {
    int i = blockIdx.x * blockDim.x + threadIdx.x;
    if (i < GRU_B * GRU_H) {
        float v0 = hidden[i];
        float v1 = hidden[GRU_B * GRU_H + i];
        h0f[i] = v0;  h0b[i] = f2bf(v0);
        h1f[i] = v1;  h1b[i] = f2bf(v1);
    }
    if (i < GRU_B) { last[i] = START_INDEX; packed[i] = 0ull; }
}

__global__ void embed_gather_kernel(const float* __restrict__ emb,
                                    const int* __restrict__ last,
                                    unsigned short* __restrict__ x_bf) {
    int b = blockIdx.x;
    long long row = (long long)last[b] * GRU_H;
    for (int k = threadIdx.x; k < GRU_H; k += blockDim.x)
        x_bf[b * GRU_H + k] = f2bf(emb[row + k]);
}

// ---- GRU cell layer: h_out = GRUCell(x, h) --------------------------------
// Block = 512 thr = 16 waves.  wave&1 -> M tile (rows 0-15 / 16-31),
// wave>>1 -> one 16-column tile of H.  Grid = H/128 = 8 blocks.
// Gates [r,z,n]: 6 wmma accumulators (3 gates x {ih,hh}), K = 1024.
__global__ __launch_bounds__(512)
void gru_layer_kernel(const unsigned short* __restrict__ x_bf,   // [32,1024]
                      const unsigned short* __restrict__ h_bf,   // [32,1024]
                      const float*          __restrict__ h_f32,  // [32,1024]
                      const unsigned short* __restrict__ w_ih,   // [3072,1024] bf16
                      const unsigned short* __restrict__ w_hh,   // [3072,1024] bf16
                      const float* __restrict__ b_ih,            // [3072]
                      const float* __restrict__ b_hh,            // [3072]
                      float*          __restrict__ h_out,        // [32,1024]
                      unsigned short* __restrict__ h_out_bf) {   // [32,1024]
    const int lane  = threadIdx.x & 31;
    const int wave  = threadIdx.x >> 5;
    const int mtile = wave & 1;
    const int ntile = wave >> 1;
    const int j     = blockIdx.x * 128 + ntile * 16 + (lane & 15);
    const int mrow  = mtile * 16 + (lane & 15);
    const int abase = (lane >> 4) * 16;
    const int bbase = (lane >> 4) * 16;

    const unsigned short* xrow = x_bf + mrow * GRU_H;
    const unsigned short* hrow = h_bf + mrow * GRU_H;
    const unsigned short* wir  = w_ih + (long long)(0 * GRU_H + j) * GRU_H;
    const unsigned short* wiz  = w_ih + (long long)(1 * GRU_H + j) * GRU_H;
    const unsigned short* win  = w_ih + (long long)(2 * GRU_H + j) * GRU_H;
    const unsigned short* whr  = w_hh + (long long)(0 * GRU_H + j) * GRU_H;
    const unsigned short* whz  = w_hh + (long long)(1 * GRU_H + j) * GRU_H;
    const unsigned short* whn  = w_hh + (long long)(2 * GRU_H + j) * GRU_H;

    v8f cir = {}, ciz = {}, cin = {}, chr_ = {}, chz = {}, chn = {};

    for (int kk = 0; kk < GRU_H; kk += 32) {
        v16bf ax = load_a_frag(xrow, kk, abase);
        v16bf ah = load_a_frag(hrow, kk, abase);
        cir  = wmma_bf16(ax, load_b_frag(wir, kk, bbase), cir);
        ciz  = wmma_bf16(ax, load_b_frag(wiz, kk, bbase), ciz);
        cin  = wmma_bf16(ax, load_b_frag(win, kk, bbase), cin);
        chr_ = wmma_bf16(ah, load_b_frag(whr, kk, bbase), chr_);
        chz  = wmma_bf16(ah, load_b_frag(whz, kk, bbase), chz);
        chn  = wmma_bf16(ah, load_b_frag(whn, kk, bbase), chn);
    }

    const float bir = b_ih[j],             bhr = b_hh[j];
    const float biz = b_ih[GRU_H + j],     bhz = b_hh[GRU_H + j];
    const float bin = b_ih[2 * GRU_H + j], bhn = b_hh[2 * GRU_H + j];

    #pragma unroll
    for (int r = 0; r < 8; ++r) {
        int m = mtile * 16 + (lane >> 4) * 8 + r;    // D layout: lane&15 = N
        float rg = sigm(cir[r] + bir + chr_[r] + bhr);
        float zg = sigm(ciz[r] + biz + chz[r] + bhz);
        float ng = tanhf(cin[r] + bin + rg * (chn[r] + bhn));
        float hp = h_f32[m * GRU_H + j];
        float hn = (1.0f - zg) * ng + zg * hp;
        h_out[m * GRU_H + j]    = hn;
        h_out_bf[m * GRU_H + j] = f2bf(hn);
    }
}

// ---- logits GEMM + bias + per-block argmax --------------------------------
// Block = 256 thr = 8 waves; each wave handles 16 vocab columns for both
// M tiles (rows 0-15 and 16-31).  Grid = 32000/128 = 250 blocks.
__global__ __launch_bounds__(256)
void logits_kernel(const unsigned short* __restrict__ h1_bf,   // [32,1024]
                   const unsigned short* __restrict__ w_out,   // [32000,1024] bf16
                   const float* __restrict__ out_b,            // [32000]
                   float* __restrict__ out,                    // [32,50,32000]
                   unsigned long long* __restrict__ packed,    // [32]
                   int t) {
    __shared__ float tile[GRU_B * 128];

    const int lane  = threadIdx.x & 31;
    const int wave  = threadIdx.x >> 5;
    const int j     = blockIdx.x * 128 + wave * 16 + (lane & 15);
    const int lcol  = wave * 16 + (lane & 15);
    const int abase = (lane >> 4) * 16;
    const int bbase = (lane >> 4) * 16;

    const unsigned short* a0row = h1_bf + (lane & 15) * GRU_H;
    const unsigned short* a1row = h1_bf + (16 + (lane & 15)) * GRU_H;
    const unsigned short* wrow  = w_out + (long long)j * GRU_H;

    v8f acc0 = {}, acc1 = {};
    for (int kk = 0; kk < GRU_H; kk += 32) {
        v16bf b  = load_b_frag(wrow, kk, bbase);
        acc0 = wmma_bf16(load_a_frag(a0row, kk, abase), b, acc0);
        acc1 = wmma_bf16(load_a_frag(a1row, kk, abase), b, acc1);
    }

    const float bias = out_b[j];
    const long long stepoff = (long long)t * VOCAB;
    #pragma unroll
    for (int r = 0; r < 8; ++r) {
        int m0 = (lane >> 4) * 8 + r;
        int m1 = 16 + m0;
        float v0 = acc0[r] + bias;
        float v1 = acc1[r] + bias;
        out[(long long)m0 * (NSTEP * (long long)VOCAB) + stepoff + j] = v0;
        out[(long long)m1 * (NSTEP * (long long)VOCAB) + stepoff + j] = v1;
        tile[m0 * 128 + lcol] = v0;
        tile[m1 * 128 + lcol] = v1;
    }
    __syncthreads();

    // per-batch argmax over this block's 128 columns, then global atomicMax
    if (threadIdx.x < GRU_B) {
        int b = threadIdx.x;
        float best = tile[b * 128];
        int   bidx = 0;
        for (int c = 1; c < 128; ++c) {
            float v = tile[b * 128 + c];
            if (v > best) { best = v; bidx = c; }     // first-max tie-break
        }
        unsigned gidx = (unsigned)(blockIdx.x * 128 + bidx);
        unsigned long long key =
            ((unsigned long long)f2ord(best) << 32) | (0xFFFFFFFFu - gidx);
        atomicMax(&packed[b], key);
    }
}

__global__ void argmax_decode_kernel(unsigned long long* __restrict__ packed,
                                     int* __restrict__ last) {
    int b = threadIdx.x;
    if (b < GRU_B) {
        last[b] = (int)(0xFFFFFFFFu - (unsigned)(packed[b] & 0xFFFFFFFFull));
        packed[b] = 0ull;                            // reset for next step
    }
}

// ---- host side ------------------------------------------------------------

extern "C" void kernel_launch(void* const* d_in, const int* in_sizes, int n_in,
                              void* d_out, int out_size, void* d_ws, size_t ws_size,
                              hipStream_t stream) {
    const float* hidden    = (const float*)d_in[0];
    const float* embedding = (const float*)d_in[1];
    const float* w_ih_l0   = (const float*)d_in[2];
    const float* w_hh_l0   = (const float*)d_in[3];
    const float* b_ih_l0   = (const float*)d_in[4];
    const float* b_hh_l0   = (const float*)d_in[5];
    const float* w_ih_l1   = (const float*)d_in[6];
    const float* w_hh_l1   = (const float*)d_in[7];
    const float* b_ih_l1   = (const float*)d_in[8];
    const float* b_hh_l1   = (const float*)d_in[9];
    const float* out_w     = (const float*)d_in[10];
    const float* out_b     = (const float*)d_in[11];
    float* out = (float*)d_out;

    // ---- workspace layout (256B aligned) ----
    const size_t WG = (size_t)3 * GRU_H * GRU_H;          // 3,145,728 elems
    const size_t WO = (size_t)VOCAB * GRU_H;              // 32,768,000 elems
    const size_t SB = (size_t)GRU_B * GRU_H;              // 32,768 elems
    char* p = (char*)d_ws;
    auto take = [&](size_t bytes) {
        char* q = p;
        p += (bytes + 255) & ~(size_t)255;
        return q;
    };
    unsigned short* wih0_bf = (unsigned short*)take(WG * 2);
    unsigned short* whh0_bf = (unsigned short*)take(WG * 2);
    unsigned short* wih1_bf = (unsigned short*)take(WG * 2);
    unsigned short* whh1_bf = (unsigned short*)take(WG * 2);
    unsigned short* outw_bf = (unsigned short*)take(WO * 2);
    unsigned short* x0_bf   = (unsigned short*)take(SB * 2);
    unsigned short* h0_bf[2] = { (unsigned short*)take(SB * 2),
                                 (unsigned short*)take(SB * 2) };
    unsigned short* h1_bf[2] = { (unsigned short*)take(SB * 2),
                                 (unsigned short*)take(SB * 2) };
    float* h0_f[2] = { (float*)take(SB * 4), (float*)take(SB * 4) };
    float* h1_f[2] = { (float*)take(SB * 4), (float*)take(SB * 4) };
    unsigned long long* packed = (unsigned long long*)take(GRU_B * 8);
    int* last = (int*)take(GRU_B * 4);
    (void)in_sizes; (void)n_in; (void)out_size; (void)ws_size;

    // ---- one-time (per call) weight conversion + state init ----
    cvt_bf16_kernel<<<2048, 256, 0, stream>>>(w_ih_l0, wih0_bf, (int)WG);
    cvt_bf16_kernel<<<2048, 256, 0, stream>>>(w_hh_l0, whh0_bf, (int)WG);
    cvt_bf16_kernel<<<2048, 256, 0, stream>>>(w_ih_l1, wih1_bf, (int)WG);
    cvt_bf16_kernel<<<2048, 256, 0, stream>>>(w_hh_l1, whh1_bf, (int)WG);
    cvt_bf16_kernel<<<4096, 256, 0, stream>>>(out_w,   outw_bf, (int)WO);
    init_state_kernel<<<(GRU_B * GRU_H + 255) / 256, 256, 0, stream>>>(
        hidden, h0_f[0], h1_f[0], h0_bf[0], h1_bf[0], last, packed);

    // ---- 50 sequential decode steps ----
    for (int t = 0; t < NSTEP; ++t) {
        const int pr = t & 1, nx = pr ^ 1;
        embed_gather_kernel<<<GRU_B, 256, 0, stream>>>(embedding, last, x0_bf);
        gru_layer_kernel<<<GRU_H / 128, 512, 0, stream>>>(
            x0_bf, h0_bf[pr], h0_f[pr], wih0_bf, whh0_bf, b_ih_l0, b_hh_l0,
            h0_f[nx], h0_bf[nx]);
        gru_layer_kernel<<<GRU_H / 128, 512, 0, stream>>>(
            h0_bf[nx], h1_bf[pr], h1_f[pr], wih1_bf, whh1_bf, b_ih_l1, b_hh_l1,
            h1_f[nx], h1_bf[nx]);
        logits_kernel<<<VOCAB / 128, 256, 0, stream>>>(
            h1_bf[nx], outw_bf, out_b, out, packed, t);
        argmax_decode_kernel<<<1, GRU_B, 0, stream>>>(packed, last);
    }
}
